// DeepGCN_51986284151369
// MI455X (gfx1250) — matrix operand order repaired and checked by reading, not compiled, not run on hardware
//
#include <hip/hip_runtime.h>
#include <math.h>

// ---------------------------------------------------------------------------
// Types for CDNA5 WMMA (wave32)
// ---------------------------------------------------------------------------
typedef __attribute__((ext_vector_type(16))) __bf16          v16bf;
typedef __attribute__((ext_vector_type(16))) unsigned short  v16us;
typedef __attribute__((ext_vector_type(8)))  float           v8f;

__device__ __forceinline__ unsigned short f2bf(float f) {
  unsigned u = __builtin_bit_cast(unsigned, f);
  u += 0x7fffu + ((u >> 16) & 1u);          // round-to-nearest-even
  return (unsigned short)(u >> 16);
}

__device__ __forceinline__ float gelu_f(float x) {
  return 0.5f * x * (1.0f + erff(x * 0.7071067811865476f)); // exact (erf) GELU
}

// ---------------------------------------------------------------------------
// WMMA GEMM on bf16 inputs: Y[m,n] = sum_k A[m,k]*B[k,n], f32 accumulate.
//   A (bf16): TRANSA ? A[k*lda+m] : A[m*lda+k]; batch stride sA (0 = shared)
//   B (bf16): B[k*ldb+n]; batch stride sB
// EPI: 0=none, 1=(y+bias)*gamma+beta, 2=EPI1+gelu, 3=EPI1+residual(f32)
// OUT: 0=f32 only, 1=bf16 only, 2=both
//
// Block = 128 threads = 4 waves; each wave owns one 16x16 tile (4 consecutive
// M tiles, shared N tile). K loop unrolled by 2 slabs of 32 (K mult of 64):
//  - !TRANSA: A slabs are layout-preserving byte copies -> staged with
//    GLOBAL_LOAD_ASYNC_TO_LDS_B128 (ASYNCcnt, one s_wait_asynccnt 0).
//  - TRANSA: bf16 ushort4 loads + b16 transpose scatter into LDS.
//  - wave 0 stages the shared B slabs transposed to [n][k] (ushort4 + b16).
//  - fragments read back as contiguous ds_load_b128 matching the
//    v_wmma_f32_16x16x32_bf16 VGPR layout.
// No edge masks: columns >= N / rows >= M carry workspace garbage, but each
// WMMA output element depends only on its own row/column, and the epilogue
// never stores out-of-range elements. All over-reads stay inside d_ws.
// Requires: K mult of 64; M mult of 16 when !TRANSA; 16B-aligned A rows when
// !TRANSA (lda mult of 8 elements).
// ---------------------------------------------------------------------------
template <bool TRANSA, int EPI, int OUT>
__global__ __launch_bounds__(128)
void wmma_gemm_kernel(const unsigned short* __restrict__ A, long long sA, int lda,
                      const unsigned short* __restrict__ Bm, long long sB, int ldb,
                      float* __restrict__ Y, unsigned short* __restrict__ Ybf,
                      long long sY, int ldy,
                      const float* __restrict__ bias,
                      const float* __restrict__ gamma,
                      const float* __restrict__ beta,
                      const float* __restrict__ Res, long long sR,
                      int M, int N, int K, int tilesN, int tilesM)
{
  __shared__ __align__(16) unsigned short LA[4][2][16][32];  // per-wave A slabs [m][k]
  __shared__ __align__(16) unsigned short LB[2][16][32];     // shared B slabs  [n][k]

  const int b    = blockIdx.y;
  const int tmg  = blockIdx.x / tilesN;
  const int tn   = blockIdx.x % tilesN;
  const int wave = threadIdx.x >> 5;
  const int lane = threadIdx.x & 31;
  int tm = tmg * 4 + wave;
  if (tm >= tilesM) tm = tilesM - 1;     // duplicate last tile: no divergent barriers

  const int m0 = tm * 16, n0 = tn * 16;
  const int lr = lane & 15, lg = lane >> 4;

  const unsigned short* Ab = A  + (long long)b * sA;
  const unsigned short* Bb = Bm + (long long)b * sB;
  const unsigned ldsA = (unsigned)(unsigned long long)&LA[wave][0][0][0];

  v8f acc = {};
  for (int k0 = 0; k0 < K; k0 += 64) {
#pragma unroll
    for (int s = 0; s < 2; ++s) {
      const int ks = k0 + s * 32;
      if (!TRANSA) {
        // A slab = 16 rows x 64B, identical layout in memory and LDS:
        // pure async byte copy, 2 x (32 lanes x 16B) = 1024B per slab.
        const int row = lane >> 2, seg = lane & 3;           // 4 x 16B per row
        const unsigned short* g0 = Ab + (long long)(m0 + row) * lda + ks + seg * 8;
        const unsigned short* g1 = g0 + (long long)8 * lda;  // rows 8..15
        const unsigned d0 = ldsA + (unsigned)(s * 1024 + (row * 32 + seg * 8) * 2);
        const unsigned d1 = d0 + 512;
        asm volatile("global_load_async_to_lds_b128 %0, %1, off" :: "v"(d0), "v"(g0) : "memory");
        asm volatile("global_load_async_to_lds_b128 %0, %1, off" :: "v"(d1), "v"(g1) : "memory");
      } else {
        // A[k,m] K-major bf16: lane = k, read 16 consecutive m, transpose scatter
        const unsigned short* src = Ab + (long long)(ks + lane) * lda + m0;
        const ushort4 u0 = *(const ushort4*)(src + 0);
        const ushort4 u1 = *(const ushort4*)(src + 4);
        const ushort4 u2 = *(const ushort4*)(src + 8);
        const ushort4 u3 = *(const ushort4*)(src + 12);
        const unsigned short vv[16] = { u0.x,u0.y,u0.z,u0.w, u1.x,u1.y,u1.z,u1.w,
                                        u2.x,u2.y,u2.z,u2.w, u3.x,u3.y,u3.z,u3.w };
#pragma unroll
        for (int j = 0; j < 16; ++j) LA[wave][s][j][lane] = vv[j];
      }
      // ---- stage shared B slab (32 K x 16 cols -> LDS [n][k]) ----
      if (wave == 0) {
        const unsigned short* src = Bb + (long long)(ks + lane) * ldb + n0;
        const ushort4 u0 = *(const ushort4*)(src + 0);
        const ushort4 u1 = *(const ushort4*)(src + 4);
        const ushort4 u2 = *(const ushort4*)(src + 8);
        const ushort4 u3 = *(const ushort4*)(src + 12);
        const unsigned short vv[16] = { u0.x,u0.y,u0.z,u0.w, u1.x,u1.y,u1.z,u1.w,
                                        u2.x,u2.y,u2.z,u2.w, u3.x,u3.y,u3.z,u3.w };
#pragma unroll
        for (int j = 0; j < 16; ++j) LB[s][j][lane] = vv[j];
      }
    }
    if (k0 + 64 < K)   // uniform branch: prefetch next B slab
      __builtin_prefetch(Bb + (long long)(k0 + 64 + lane) * ldb + n0, 0, 1);

    if (!TRANSA)
      asm volatile("s_wait_asynccnt 0x0" ::: "memory");    // own A slabs landed
    __syncthreads();

    // ---- fragment reads (match WMMA bf16 VGPR layout) + 2 WMMAs ----
#pragma unroll
    for (int s = 0; s < 2; ++s) {
      union { uint4 q[2]; v16us u; } fa, fb;
      fa.q[0] = *(const uint4*)&LA[wave][s][lr][lg * 8];
      fa.q[1] = *(const uint4*)&LA[wave][s][lr][16 + lg * 8];
      fb.q[0] = *(const uint4*)&LB[s][lr][lg * 8];
      fb.q[1] = *(const uint4*)&LB[s][lr][16 + lg * 8];
      acc = __builtin_amdgcn_wmma_f32_16x16x32_bf16(
          false, __builtin_bit_cast(v16bf, fa.u),
          false, __builtin_bit_cast(v16bf, fb.u),
          (short)0, acc, false, false);
    }
    __syncthreads();
  }

  float*          Yb  = Y   + (long long)b * sY;
  unsigned short* Ybb = Ybf + (long long)b * sY;
  const float*    Rb  = (EPI == 3) ? (Res + (long long)b * sR) : nullptr;
#pragma unroll
  for (int r = 0; r < 8; ++r) {
    const int m = m0 + r + lg * 8;        // C/D layout: lanes 16-31 hold M=r+8
    const int n = n0 + lr;
    if (m < M && n < N) {
      float v = acc[r];
      if (EPI >= 1) v = (v + bias[m]) * gamma[m] + beta[m];
      if (EPI == 2) v = gelu_f(v);
      if (EPI == 3) v += Rb[(long long)m * ldy + n];
      if (OUT != 1) Yb [(long long)m * ldy + n] = v;
      if (OUT >= 1) Ybb[(long long)m * ldy + n] = f2bf(v);
    }
  }
}

// ---------------------------------------------------------------------------
// One-time fp32 -> bf16 weight conversion
// ---------------------------------------------------------------------------
__global__ void cvt_bf16_kernel(const float* __restrict__ s,
                                unsigned short* __restrict__ d, int n)
{
  const int t = blockIdx.x * blockDim.x + threadIdx.x;
  if (t < n) d[t] = f2bf(s[t]);
}

// ---------------------------------------------------------------------------
// Stem: direct 3x3 conv, pad 1, fused eval-BN (+ optional GELU)
// ---------------------------------------------------------------------------
template <bool ACT>
__global__ void conv3x3_bn_kernel(const float* __restrict__ X,
                                  const float* __restrict__ Wt,
                                  const float* __restrict__ bias,
                                  const float* __restrict__ gam,
                                  const float* __restrict__ bet,
                                  float* __restrict__ Y,
                                  int Bn, int Cin, int Hin, int Win,
                                  int Cout, int Hout, int Wout, int stride)
{
  long long idx = (long long)blockIdx.x * blockDim.x + threadIdx.x;
  const long long total = (long long)Bn * Cout * Hout * Wout;
  if (idx >= total) return;
  const int ow = (int)(idx % Wout); long long t = idx / Wout;
  const int oh = (int)(t % Hout);   t /= Hout;
  const int oc = (int)(t % Cout);
  const int b  = (int)(t / Cout);

  const float* xb = X  + (long long)b * Cin * Hin * Win;
  const float* wf = Wt + (long long)oc * Cin * 9;
  float acc = 0.0f;
  for (int ic = 0; ic < Cin; ++ic) {
    const float* xc = xb + (long long)ic * Hin * Win;
    const float* wc = wf + ic * 9;
#pragma unroll
    for (int kh = 0; kh < 3; ++kh) {
      const int ih = oh * stride + kh - 1;
      if (ih < 0 || ih >= Hin) continue;
#pragma unroll
      for (int kw = 0; kw < 3; ++kw) {
        const int iw = ow * stride + kw - 1;
        if (iw < 0 || iw >= Win) continue;
        acc += xc[(long long)ih * Win + iw] * wc[kh * 3 + kw];
      }
    }
  }
  float v = (acc + bias[oc]) * gam[oc] + bet[oc];
  if (ACT) v = gelu_f(v);
  Y[idx] = v;
}

// pos-embed add; writes fp32 XT and bf16 copy for the first GEMM
__global__ void posadd_kernel(float* __restrict__ X, unsigned short* __restrict__ Xbf,
                              const float* __restrict__ pos, int Bn, int CN)
{
  long long t = (long long)blockIdx.x * blockDim.x + threadIdx.x;
  if (t >= (long long)Bn * CN) return;
  const float v = X[t] + pos[t % CN];
  X[t] = v;
  Xbf[t] = f2bf(v);
}

// L2-normalize each token column of H (B,C,N) -> bf16 XN (only GEMM consumes it)
__global__ void normalize_kernel(const float* __restrict__ H,
                                 unsigned short* __restrict__ XN,
                                 int Bn, int C, int N)
{
  const int t = blockIdx.x * blockDim.x + threadIdx.x;
  if (t >= Bn * N) return;
  const int b = t / N, n = t % N;
  const float* base = H + (long long)b * C * N + n;
  float s = 0.0f;
  for (int c = 0; c < C; ++c) { const float v = base[(long long)c * N]; s += v * v; }
  const float inv = 1.0f / fmaxf(sqrtf(s), 1e-12f);
  unsigned short* o = XN + (long long)b * C * N + n;
  for (int c = 0; c < C; ++c) o[(long long)c * N] = f2bf(base[(long long)c * N] * inv);
}

// top-k by descending similarity (stable: ascending index tie-break)
__global__ void topk_kernel(const float* __restrict__ S, int* __restrict__ IDX,
                            int Bn, int N, int k)
{
  const int t = blockIdx.x * blockDim.x + threadIdx.x;
  if (t >= Bn * N) return;
  const int b = t / N, n = t % N;
  const float* row = S + ((long long)b * N + n) * N;
  int* out = IDX + ((long long)b * N + n) * k;
  float pv = 3.4e38f; int pj = -1;
  for (int s = 0; s < k; ++s) {
    float best = -3.4e38f; int bj = 0;
    for (int j = 0; j < N; ++j) {
      const float v = row[j];
      const bool elig = (v < pv) || (v == pv && j > pj);
      if (elig && v > best) { best = v; bj = j; }
    }
    out[s] = bj; pv = best; pj = bj;
  }
}

// max-relative graph conv gather -> bf16 (only the cw GEMM consumes it)
__global__ void mrconv_kernel(const float* __restrict__ H, const int* __restrict__ IDX,
                              unsigned short* __restrict__ Y, int Bn, int C, int N, int k)
{
  long long t = (long long)blockIdx.x * blockDim.x + threadIdx.x;
  if (t >= (long long)Bn * C * N) return;
  const int n = (int)(t % N); long long r = t / N;
  const int c = (int)(r % C);
  const int b = (int)(r / C);
  const float* hb = H + (long long)b * C * N + (long long)c * N;
  const float  x  = hb[n];
  const int* id = IDX + ((long long)b * N + n) * k;
  float m = -3.4e38f;
  for (int s = 0; s < k; ++s) m = fmaxf(m, hb[id[s]] - x);
  unsigned short* yb = Y + (long long)b * (2 * C) * N;
  yb[(long long)(2 * c)     * N + n] = f2bf(x);
  yb[(long long)(2 * c + 1) * N + n] = f2bf(m);
}

__global__ void pool_kernel(const float* __restrict__ X, float* __restrict__ P,
                            int Bn, int C, int N)
{
  const int t = blockIdx.x * blockDim.x + threadIdx.x;
  if (t >= Bn * C) return;
  const int b = t / C, c = t % C;
  const float* x = X + (long long)b * C * N + (long long)c * N;
  float s = 0.0f;
  for (int n = 0; n < N; ++n) s += x[n];
  P[t] = s / (float)N;
}

__global__ void head1_kernel(const float* __restrict__ P, const float* __restrict__ W,
                             const float* __restrict__ bias, const float* __restrict__ g,
                             const float* __restrict__ bb, float* __restrict__ O,
                             int Bn, int Cin, int Cout)
{
  const int t = blockIdx.x * blockDim.x + threadIdx.x;
  if (t >= Bn * Cout) return;
  const int b = t / Cout, o = t % Cout;
  const float* p = P + (long long)b * Cin;
  const float* w = W + (long long)o * Cin;
  float a = 0.0f;
  for (int c = 0; c < Cin; ++c) a += w[c] * p[c];
  O[t] = gelu_f((a + bias[o]) * g[o] + bb[o]);
}

__global__ void head2_kernel(const float* __restrict__ P, const float* __restrict__ W,
                             const float* __restrict__ bias, float* __restrict__ O,
                             int Bn, int Cin, int Cout)
{
  const int t = blockIdx.x * blockDim.x + threadIdx.x;
  if (t >= Bn * Cout) return;
  const int b = t / Cout, o = t % Cout;
  const float* p = P + (long long)b * Cin;
  const float* w = W + (long long)o * Cin;
  float a = 0.0f;
  for (int c = 0; c < Cin; ++c) a += w[c] * p[c];
  O[t] = a + bias[o];
}

// ---------------------------------------------------------------------------
// Launcher
// ---------------------------------------------------------------------------
static inline int tilesOf(int x) { return (x + 15) / 16; }

extern "C" void kernel_launch(void* const* d_in, const int* in_sizes, int n_in,
                              void* d_out, int out_size, void* d_ws, size_t ws_size,
                              hipStream_t stream)
{
  (void)in_sizes; (void)n_in; (void)out_size; (void)ws_size;
  constexpr int B = 32, C = 192, N = 196;
  static const int NUM_KNN[12] = {9, 9, 10, 11, 12, 13, 13, 14, 15, 16, 17, 18};

  // ---- unpack inputs (dict order of setup_inputs, depth-first) ----
  int pi = 0;
  const float* inp = (const float*)d_in[pi++];                // (32,3,224,224)
  struct StemP { const float *w, *b, *g, *bb; } stem[5];
  for (int i = 0; i < 5; ++i) {
    stem[i] = { (const float*)d_in[pi], (const float*)d_in[pi + 1],
                (const float*)d_in[pi + 2], (const float*)d_in[pi + 3] };
    pi += 4;
  }
  const float* pos = (const float*)d_in[pi++];                // (1,192,14,14)
  struct BlkP {
    const float *f1w, *f1b, *cw, *cb, *f2w, *f2b;
    const float *f1g, *f1bb, *cg, *cbb, *f2g, *f2bb;
    const float *w1, *b1, *w2, *b2, *g1, *bb1, *g2, *bb2;
  } blk[12];
  for (int i = 0; i < 12; ++i) {
    BlkP& p = blk[i];
    p.f1w = (const float*)d_in[pi++]; p.f1b = (const float*)d_in[pi++];
    p.cw  = (const float*)d_in[pi++]; p.cb  = (const float*)d_in[pi++];
    p.f2w = (const float*)d_in[pi++]; p.f2b = (const float*)d_in[pi++];
    p.f1g = (const float*)d_in[pi++]; p.f1bb = (const float*)d_in[pi++];
    p.cg  = (const float*)d_in[pi++]; p.cbb  = (const float*)d_in[pi++];
    p.f2g = (const float*)d_in[pi++]; p.f2bb = (const float*)d_in[pi++];
    p.w1 = (const float*)d_in[pi++]; p.b1 = (const float*)d_in[pi++];
    p.w2 = (const float*)d_in[pi++]; p.b2 = (const float*)d_in[pi++];
    p.g1 = (const float*)d_in[pi++]; p.bb1 = (const float*)d_in[pi++];
    p.g2 = (const float*)d_in[pi++]; p.bb2 = (const float*)d_in[pi++];
  }
  const float* pw1  = (const float*)d_in[pi++];
  const float* pb1  = (const float*)d_in[pi++];
  const float* pg1  = (const float*)d_in[pi++];
  const float* pbb1 = (const float*)d_in[pi++];
  const float* pw2  = (const float*)d_in[pi++];
  const float* pb2  = (const float*)d_in[pi++];

  // ---- workspace layout (float units unless noted) ----
  float* ws = (float*)d_ws;
  float* R1 = ws;                         // 9,633,792  stem conv1 | token: S + H
  float* R2 = R1 + 9633792;               // 8,429,568  stem conv2/3/4 | token: bf16 acts
  float* XT = R2 + 8429568;               // 1,204,224  persistent fp32 tokens
  float* WB = XT + 1204224;               // 3,317,760  float-equiv: bf16 weights
  float* EX = WB + 3317760;               // idx + pool + head scratch
  // stem aliases
  float* A2s = R2;                        // 32*48*56*56   = 4,816,896
  float* A3s = R2 + 4816896;              // 32*96*28*28   = 2,408,448
  float* A4s = R2 + 7225344;              // 32*192*14*14  = 1,204,224
  // token-phase fp32 aliases (inside R1)
  float* S = R1;                          // 32*196*196 = 1,229,312
  float* H = R1 + 1229312;                // 32*192*196 = 1,204,224
  // token-phase bf16 activations (inside R2, ushort units)
  unsigned short* XTbf   = (unsigned short*)R2;
  unsigned short* XNbf   = XTbf + 1204224;
  unsigned short* Y384bf = XNbf + 1204224;      // 2,408,448
  unsigned short* Y384Bbf= Y384bf + 2408448;    // 2,408,448
  unsigned short* H768bf = Y384Bbf + 2408448;   // 4,816,896 (slack after, in R2)
  // bf16 weights (ushort units)
  unsigned short* WBbf = (unsigned short*)WB;   // 12 * 552,960
  int*   IDXb  = (int*)EX;                      // 32*196*18
  float* POOL  = EX + 112896;
  float* HEADB = POOL + 6144;

  const long long sTok = (long long)C * N;
  const long long s384 = (long long)2 * C * N;
  const long long s768 = (long long)4 * C * N;
  const long long sS   = (long long)N * N;

  auto cdiv = [](long long a, int b2) { return (int)((a + b2 - 1) / b2); };
  auto gemmGrid = [&](int tilesM, int tilesN2) {
    return dim3((unsigned)(((tilesM + 3) / 4) * tilesN2), (unsigned)B);
  };

  // ---- one-time weight conversion to bf16 ----
  unsigned short* wbf[12][5];
  {
    const int wsz[5] = { C * C, 2 * C * 2 * C, C * 2 * C, 4 * C * C, C * 4 * C };
    unsigned short* cur = WBbf;
    for (int i = 0; i < 12; ++i) {
      const float* srcs[5] = { blk[i].f1w, blk[i].cw, blk[i].f2w, blk[i].w1, blk[i].w2 };
      for (int j = 0; j < 5; ++j) {
        wbf[i][j] = cur;
        cvt_bf16_kernel<<<cdiv(wsz[j], 256), 256, 0, stream>>>(srcs[j], cur, wsz[j]);
        cur += wsz[j];
      }
    }
  }

  // ---- stem ----
  {
    const int dims[6][3] = { {3,224,224}, {24,112,112}, {48,56,56},
                             {96,28,28}, {192,14,14}, {192,14,14} };
    const int strides[5] = {2, 2, 2, 2, 1};
    const float* src = inp;
    float* dst[5] = { R1, A2s, A3s, A4s, XT };
    for (int i = 0; i < 5; ++i) {
      const long long tot = (long long)B * dims[i + 1][0] * dims[i + 1][1] * dims[i + 1][2];
      const dim3 g(cdiv(tot, 256));
      if (i < 4)
        conv3x3_bn_kernel<true><<<g, 256, 0, stream>>>(
            src, stem[i].w, stem[i].b, stem[i].g, stem[i].bb, dst[i],
            B, dims[i][0], dims[i][1], dims[i][2],
            dims[i + 1][0], dims[i + 1][1], dims[i + 1][2], strides[i]);
      else
        conv3x3_bn_kernel<false><<<g, 256, 0, stream>>>(
            src, stem[i].w, stem[i].b, stem[i].g, stem[i].bb, dst[i],
            B, dims[i][0], dims[i][1], dims[i][2],
            dims[i + 1][0], dims[i + 1][1], dims[i + 1][2], strides[i]);
      src = dst[i];
    }
    posadd_kernel<<<cdiv((long long)B * sTok, 256), 256, 0, stream>>>(
        XT, XTbf, pos, B, (int)sTok);
  }

  const int tN = tilesOf(N);   // 13

  // ---- 12 Grapher + FFN blocks ----
  for (int i = 0; i < 12; ++i) {
    const BlkP& p = blk[i];
    const int k = NUM_KNN[i];

    // f1: H = BN(W1 x + b)   (f32 out only)
    wmma_gemm_kernel<false, 1, 0><<<gemmGrid(tilesOf(C), tN), 128, 0, stream>>>(
        wbf[i][0], 0, C, XTbf, sTok, N, H, nullptr, sTok, N,
        p.f1b, p.f1g, p.f1bb, nullptr, 0, C, N, C, tN, tilesOf(C));

    // KNN: normalize -> bf16, Gram matrix (WMMA, A transposed), top-k
    normalize_kernel<<<cdiv(B * N, 256), 256, 0, stream>>>(H, XNbf, B, C, N);
    wmma_gemm_kernel<true, 0, 0><<<gemmGrid(tN, tN), 128, 0, stream>>>(
        XNbf, sTok, N, XNbf, sTok, N, S, nullptr, sS, N,
        nullptr, nullptr, nullptr, nullptr, 0, N, N, C, tN, tN);
    topk_kernel<<<cdiv(B * N, 256), 256, 0, stream>>>(S, IDXb, B, N, k);

    // gather -> bf16 (B,384,N); cw GEMM (+GELU, bf16 out); f2 (+residual, dual out)
    mrconv_kernel<<<cdiv((long long)B * C * N, 256), 256, 0, stream>>>(
        H, IDXb, Y384bf, B, C, N, k);
    wmma_gemm_kernel<false, 2, 1><<<gemmGrid(tilesOf(2 * C), tN), 128, 0, stream>>>(
        wbf[i][1], 0, 2 * C, Y384bf, s384, N, nullptr, Y384Bbf, s384, N,
        p.cb, p.cg, p.cbb, nullptr, 0, 2 * C, N, 2 * C, tN, tilesOf(2 * C));
    wmma_gemm_kernel<false, 3, 2><<<gemmGrid(tilesOf(C), tN), 128, 0, stream>>>(
        wbf[i][2], 0, 2 * C, Y384Bbf, s384, N, XT, XTbf, sTok, N,
        p.f2b, p.f2g, p.f2bb, XT, sTok, C, N, 2 * C, tN, tilesOf(C));

    // FFN: 192 -> 768 (GELU, bf16 out) -> 192 (+residual, dual out)
    wmma_gemm_kernel<false, 2, 1><<<gemmGrid(tilesOf(4 * C), tN), 128, 0, stream>>>(
        wbf[i][3], 0, C, XTbf, sTok, N, nullptr, H768bf, s768, N,
        p.b1, p.g1, p.bb1, nullptr, 0, 4 * C, N, C, tN, tilesOf(4 * C));
    wmma_gemm_kernel<false, 3, 2><<<gemmGrid(tilesOf(C), tN), 128, 0, stream>>>(
        wbf[i][4], 0, 4 * C, H768bf, s768, N, XT, XTbf, sTok, N,
        p.b2, p.g2, p.bb2, XT, sTok, C, N, 4 * C, tN, tilesOf(C));
  }

  // ---- head: mean pool -> 1024 (BN+GELU) -> 10 logits ----
  pool_kernel<<<cdiv(B * C, 256), 256, 0, stream>>>(XT, POOL, B, C, N);
  head1_kernel<<<cdiv(B * 1024, 256), 256, 0, stream>>>(
      POOL, pw1, pb1, pg1, pbb1, HEADB, B, C, 1024);
  head2_kernel<<<cdiv(B * 10, 64), 64, 0, stream>>>(
      HEADB, pw2, pb2, (float*)d_out, B, 1024, 10);
}
